// VAEDecoder_49495203119742
// MI455X (gfx1250) — matrix operand order repaired and checked by reading, not compiled, not run on hardware
//
#include <hip/hip_runtime.h>
#include <hip/hip_bf16.h>

// Problem constants (match reference)
#define B_   512
#define LAT_ 128
#define H_   1024
#define T_   256
#define NBLKS 32     // H_/32 n-blocks per step kernel

typedef __attribute__((ext_vector_type(16))) __bf16      v16bf;
typedef __attribute__((ext_vector_type(8)))  float       v8f;
typedef __attribute__((ext_vector_type(4)))  unsigned int u32x4;

union FragBF { v16bf v; u32x4 u[2]; };

__device__ __forceinline__ float sigmoidf_(float x) { return 1.0f / (1.0f + __expf(-x)); }

// ---------------------------------------------------------------------------
// Prep 1: swizzle W_hh (4096x1024 f32, row-major) into bf16 WMMA B-fragments.
// Fragment (jtile, kc): lane holds n = lane&15 (col j = jtile*16+n),
// elements e=0..15 hold k = kc*32 + (lane>>4)*16 + e. Stored contiguously:
// 32 bytes per lane -> two fully-coalesced b128 loads in the step kernel.
// ---------------------------------------------------------------------------
__global__ void k_swizzle_whh(const float* __restrict__ Whh, __bf16* __restrict__ wfrag) {
    int gid  = blockIdx.x * blockDim.x + threadIdx.x;   // frag*32 + lane
    int lane = gid & 31;
    int frag = gid >> 5;                                // 0..8191
    int kc   = frag & 31;
    int jt   = frag >> 5;                               // 0..255
    int j    = jt * 16 + (lane & 15);
    int kb   = kc * 32 + (lane >> 4) * 16;
    const float* src = Whh + (size_t)j * H_ + kb;
    FragBF f;
#pragma unroll
    for (int e = 0; e < 16; ++e) f.v[e] = (__bf16)src[e];
    u32x4* dst = (u32x4*)(wfrag + (size_t)gid * 16);
    dst[0] = f.u[0];
    dst[1] = f.u[1];
}

// ---------------------------------------------------------------------------
// Prep 2: initial states h0 = tanh(z@Wzh^T + bzh) (bf16), c0 = tanh(z@Wzc^T + bzc)
// ---------------------------------------------------------------------------
__global__ void k_init_state(const float* __restrict__ z,
                             const float* __restrict__ Wzh, const float* __restrict__ bzh,
                             const float* __restrict__ Wzc, const float* __restrict__ bzc,
                             __bf16* __restrict__ h0, float* __restrict__ c0) {
    int idx = blockIdx.x * blockDim.x + threadIdx.x;    // b*H + n
    if (idx >= B_ * H_) return;
    int b = idx >> 10;
    int n = idx & (H_ - 1);
    const float* zr = z   + (size_t)b * LAT_;
    const float* wh = Wzh + (size_t)n * LAT_;
    const float* wc = Wzc + (size_t)n * LAT_;
    float sh = 0.f, sc = 0.f;
#pragma unroll 4
    for (int k = 0; k < LAT_; ++k) { sh += zr[k] * wh[k]; sc += zr[k] * wc[k]; }
    h0[idx] = (__bf16)tanhf(sh + bzh[n]);
    c0[idx] = tanhf(sc + bzc[n]);
}

// ---------------------------------------------------------------------------
// One LSTM step, fully fused. Workgroup tile: 64 batch rows x 32 n-cols,
// all 4 gates. 8 waves; wave w: gate g=w&3, batch-half mh=w>>2.
// Each wave: 2x2 register tile (2 M-subtiles x 2 N-tiles, same gate) ->
// 4 WMMAs per K-chunk on 2 A-frag + 2 B-frag loads (2 b128 loads per WMMA).
// ---------------------------------------------------------------------------
__global__ void __launch_bounds__(256, 2)
k_lstm_step(const __bf16* __restrict__ hprev,   // [B_][H_] bf16 row-major
            __bf16*       __restrict__ hnext,
            float*        __restrict__ cst,     // [B_][H_] f32
            const __bf16* __restrict__ wfrag,   // swizzled W_hh
            const float*  __restrict__ W_ih,    // [4H]
            const float*  __restrict__ b_ih,
            const float*  __restrict__ b_hh,
            const float*  __restrict__ W_out,   // [H]
            const float*  __restrict__ out,     // read x = out[b*T + t-1]
            float*        __restrict__ ypart,   // [NBLKS][B_] split-k partials
            int t) {
    __shared__ float gl[4][64][33];             // gates, padded vs bank conflicts
    __shared__ float yred[4][64];

    const int nblock = blockIdx.x;              // 0..31 (32-wide n-block within H)
    const int mblock = blockIdx.y;              // 0..7  (64 batch rows each)
    const int tid    = threadIdx.x;
    const int lane   = tid & 31;
    const int w      = tid >> 5;
    const int g      = w & 3;                   // gate
    const int mh     = w >> 2;                  // 0/1 (which pair of M-subtiles)
    const int nloc   = lane & 15;
    const int khalf  = lane >> 4;

    v8f acc00 = {0.f,0.f,0.f,0.f,0.f,0.f,0.f,0.f};
    v8f acc01 = acc00, acc10 = acc00, acc11 = acc00;

    // A-operand rows (M = lane&15 within each 16-row subtile)
    const int m0 = mblock * 64 + (mh * 2 + 0) * 16 + nloc;
    const int m1 = mblock * 64 + (mh * 2 + 1) * 16 + nloc;

    // Column tiles in the 4096-wide gate matrix for the two N-tiles
    const int jt0 = g * 64 + nblock * 2 + 0;
    const int jt1 = g * 64 + nblock * 2 + 1;

    const u32x4* aptr = (const u32x4*)hprev;
    const u32x4* bptr = (const u32x4*)wfrag;

    // A: byte = m*2048 + kc*64 + khalf*32  -> u32x4 index m*128 + kc*4 + khalf*2 (+1)
    size_t ai0 = (size_t)m0 * 128 + khalf * 2;
    size_t ai1 = (size_t)m1 * 128 + khalf * 2;
    // B: u32x4 index = ((jtile*32 + kc)*32 + lane)*2 (+1)
    size_t bi0 = ((size_t)jt0 * 1024 + lane) * 2;
    size_t bi1 = ((size_t)jt1 * 1024 + lane) * 2;

    for (int kc = 0; kc < 32; ++kc) {
        FragBF fa0, fa1, fb0, fb1;
        fa0.u[0] = aptr[ai0];     fa0.u[1] = aptr[ai0 + 1];
        fa1.u[0] = aptr[ai1];     fa1.u[1] = aptr[ai1 + 1];
        fb0.u[0] = bptr[bi0];     fb0.u[1] = bptr[bi0 + 1];
        fb1.u[0] = bptr[bi1];     fb1.u[1] = bptr[bi1 + 1];
        acc00 = __builtin_amdgcn_wmma_f32_16x16x32_bf16(false, fa0.v, false, fb0.v,
                                                        (short)0, acc00, false, false);
        acc01 = __builtin_amdgcn_wmma_f32_16x16x32_bf16(false, fa0.v, false, fb1.v,
                                                        (short)0, acc01, false, false);
        acc10 = __builtin_amdgcn_wmma_f32_16x16x32_bf16(false, fa1.v, false, fb0.v,
                                                        (short)0, acc10, false, false);
        acc11 = __builtin_amdgcn_wmma_f32_16x16x32_bf16(false, fa1.v, false, fb1.v,
                                                        (short)0, acc11, false, false);
        ai0 += 4; ai1 += 4; bi0 += 64; bi1 += 64;
    }

    // Epilogue stage 1: add bias + x*W_ih, exchange gates through LDS.
    const int   j0 = g * H_ + nblock * 32 + 0  + nloc;
    const int   j1 = g * H_ + nblock * 32 + 16 + nloc;
    const float bias0 = b_ih[j0] + b_hh[j0];
    const float bias1 = b_ih[j1] + b_hh[j1];
    const float wih0  = W_ih[j0];
    const float wih1  = W_ih[j1];
#pragma unroll
    for (int r = 0; r < 8; ++r) {
        // C-layout: element r -> row (r + 8*khalf) within the 16-row subtile
        int mi0 = (mh * 2 + 0) * 16 + khalf * 8 + r;
        int mi1 = (mh * 2 + 1) * 16 + khalf * 8 + r;
        int bg0 = mblock * 64 + mi0;
        int bg1 = mblock * 64 + mi1;
        float x0 = (t > 0) ? out[(size_t)bg0 * T_ + (t - 1)] : 0.f;
        float x1 = (t > 0) ? out[(size_t)bg1 * T_ + (t - 1)] : 0.f;
        gl[g][mi0][nloc]      = acc00[r] + x0 * wih0 + bias0;
        gl[g][mi0][16 + nloc] = acc01[r] + x0 * wih1 + bias1;
        gl[g][mi1][nloc]      = acc10[r] + x1 * wih0 + bias0;
        gl[g][mi1][16 + nloc] = acc11[r] + x1 * wih1 + bias1;
    }
    __syncthreads();

    // Epilogue stage 2: nonlinearities, state update, partial y.
    const int row = tid & 63;                   // row within 64-batch block
    const int cg  = tid >> 6;                   // 0..3, 8 columns each
    const int bg  = mblock * 64 + row;
    float ysum = 0.f;
#pragma unroll
    for (int q = 0; q < 8; ++q) {
        int col  = cg * 8 + q;
        int nidx = nblock * 32 + col;
        float gi = sigmoidf_(gl[0][row][col]);
        float gf = sigmoidf_(gl[1][row][col]);
        float gg = tanhf    (gl[2][row][col]);
        float go = sigmoidf_(gl[3][row][col]);
        size_t sidx = (size_t)bg * H_ + nidx;
        float cn = gf * cst[sidx] + gi * gg;
        float hn = go * tanhf(cn);
        cst[sidx]   = cn;
        hnext[sidx] = (__bf16)hn;
        ysum += hn * W_out[nidx];
    }
    yred[cg][row] = ysum;
    __syncthreads();
    if (tid < 64) {
        float yy = yred[0][tid] + yred[1][tid] + yred[2][tid] + yred[3][tid];
        ypart[(size_t)nblock * B_ + mblock * 64 + tid] = yy;  // deterministic split-k
    }
}

// ---------------------------------------------------------------------------
// Per-step reduce: y[b] = b_out + sum over n-block partials; feeds next x.
// ---------------------------------------------------------------------------
__global__ void k_reduce_y(const float* __restrict__ ypart,
                           const float* __restrict__ b_out,
                           float* __restrict__ out, int t) {
    int b = blockIdx.x * blockDim.x + threadIdx.x;
    if (b >= B_) return;
    float s = b_out[0];
#pragma unroll 8
    for (int nt = 0; nt < NBLKS; ++nt) s += ypart[(size_t)nt * B_ + b];
    out[(size_t)b * T_ + t] = s;
}

// ---------------------------------------------------------------------------
extern "C" void kernel_launch(void* const* d_in, const int* in_sizes, int n_in,
                              void* d_out, int out_size, void* d_ws, size_t ws_size,
                              hipStream_t stream) {
    (void)in_sizes; (void)n_in; (void)out_size; (void)ws_size;
    const float* z     = (const float*)d_in[0];
    const float* Wzh   = (const float*)d_in[1];
    const float* bzh   = (const float*)d_in[2];
    const float* Wzc   = (const float*)d_in[3];
    const float* bzc   = (const float*)d_in[4];
    const float* W_ih  = (const float*)d_in[5];
    const float* W_hh  = (const float*)d_in[6];
    const float* b_ih  = (const float*)d_in[7];
    const float* b_hh  = (const float*)d_in[8];
    const float* W_out = (const float*)d_in[9];
    const float* b_out = (const float*)d_in[10];
    float* out = (float*)d_out;

    // Workspace layout (re-initialized every call; never relies on old state):
    //   [0,   8MB)  W_hh bf16 fragments
    //   [8MB, 9MB)  h buffer 0 (bf16)
    //   [9MB,10MB)  h buffer 1 (bf16)
    //   [10MB,12MB) c state (f32)
    //   [12MB, +64KB) split-k y partials
    char* ws = (char*)d_ws;
    __bf16* wfrag = (__bf16*)(ws);
    __bf16* hbuf0 = (__bf16*)(ws + (size_t)(8u  << 20));
    __bf16* hbuf1 = (__bf16*)(ws + (size_t)(9u  << 20));
    float*  cst   = (float*) (ws + (size_t)(10u << 20));
    float*  ypart = (float*) (ws + (size_t)(12u << 20));

    k_swizzle_whh<<<(8192 * 32) / 256, 256, 0, stream>>>(W_hh, wfrag);
    k_init_state<<<(B_ * H_) / 256, 256, 0, stream>>>(z, Wzh, bzh, Wzc, bzc, hbuf0, cst);

    dim3 grid(NBLKS, 8);
    for (int t = 0; t < T_; ++t) {
        __bf16* hp = (t & 1) ? hbuf1 : hbuf0;
        __bf16* hn = (t & 1) ? hbuf0 : hbuf1;
        k_lstm_step<<<grid, 256, 0, stream>>>(hp, hn, cst, wfrag,
                                              W_ih, b_ih, b_hh, W_out,
                                              out, ypart, t);
        k_reduce_y<<<(B_ + 255) / 256, 256, 0, stream>>>(ypart, b_out, out, t);
    }
}
